// CGNN_88983132438708
// MI455X (gfx1250) — matrix operand order repaired and checked by reading, not compiled, not run on hardware
//
#include <hip/hip_runtime.h>
#include <hip/hip_bf16.h>
#include <math.h>

// ---------------------------------------------------------------------------
// CGNN pipeline for MI455X (gfx1250, wave32, WMMA):
//   12 x [8192x8192]@[8192x128] GEMMs dominate (~206 GFLOP). adj converted to
//   bf16 once (128MB -> L2-resident on the 192MB L2), computed with
//   v_wmma_f32_16x16x32_bf16. h kept bf16-TRANSPOSED so B fragments and C
//   stores are contiguous 16B accesses.
// GEMM v2: 32 output rows per block (2 row-strips/wave sharing each B frag ->
//   2x wmma per B load, halved L2 B-traffic) + double-buffered LDS A staging
//   (A-strip global load overlaps the 8 WMMAs of the current stage; one
//   barrier per stage).
// ---------------------------------------------------------------------------

#define NN 8192
#define DIN 256
#define HID 128
#define DOUT 64

typedef unsigned short u16;
typedef __attribute__((ext_vector_type(16))) __bf16 v16bf;
typedef __attribute__((ext_vector_type(8)))  float  v8f;
typedef __attribute__((ext_vector_type(8)))  unsigned short us8;
typedef __attribute__((ext_vector_type(4)))  unsigned short us4;
typedef __attribute__((ext_vector_type(4)))  float  f4;

union BF16x16 { struct { us8 lo, hi; } h; v16bf v; };

__device__ __forceinline__ u16 f2bf(float f) {
    union { float f; unsigned u; } c; c.f = f;
    unsigned u = c.u;
    return (u16)((u + 0x7FFFu + ((u >> 16) & 1u)) >> 16);   // RNE
}
__device__ __forceinline__ float bf2f(u16 h) {
    union { unsigned u; float f; } c; c.u = ((unsigned)h) << 16;
    return c.f;
}

// ---------------------------------------------------------------------------
// adj fp32 -> bf16 (one pass over 256MB; makes adj L2-resident at 128MB)
// ---------------------------------------------------------------------------
__global__ __launch_bounds__(256) void k_conv_adj(const float* __restrict__ adj,
                                                  u16* __restrict__ adjB) {
    size_t i = ((size_t)blockIdx.x * 256 + threadIdx.x) * 4;
    f4 v = *(const f4*)(adj + i);
    us4 o;
    o.x = f2bf(v.x); o.y = f2bf(v.y); o.z = f2bf(v.z); o.w = f2bf(v.w);
    *(us4*)(adjB + i) = o;
}

// ---------------------------------------------------------------------------
// epilogue helper: write one 16x16 tile's 8-row column slice (contiguous 16B)
// ---------------------------------------------------------------------------
__device__ __forceinline__ void epi_store(const v8f& acc, u16* __restrict__ Outt,
                                          const u16* __restrict__ Oldt,
                                          size_t idx, int mode, float c0, float c1) {
    us8 ov;
    if (mode == 2) {                              // ODE: h' = c0*h + c1*(A h)
        us8 oldv = *(const us8*)(Oldt + idx);
#pragma unroll
        for (int r = 0; r < 8; ++r)
            ov[r] = f2bf(c0 * bf2f(oldv[r]) + c1 * acc[r]);
    } else if (mode == 1) {                       // relu
#pragma unroll
        for (int r = 0; r < 8; ++r)
            ov[r] = f2bf(fmaxf(acc[r], 0.0f));
    } else {                                      // copy
#pragma unroll
        for (int r = 0; r < 8; ++r)
            ov[r] = f2bf(acc[r]);
    }
    *(us8*)(Outt + idx) = ov;
}

// ---------------------------------------------------------------------------
// Big GEMM: Out^T = epilogue( adj @ B ), B given bf16-transposed [HID][NN].
// Block: 32 output rows x 128 cols, 8 waves. Wave = 1 col-tile x 2 row-strips
// (2 accumulators share each B fragment). A staged via double-buffered LDS.
// ---------------------------------------------------------------------------
__global__ __launch_bounds__(256) void k_gemm_adj(const u16* __restrict__ adjB,
                                                  const u16* __restrict__ Bt,
                                                  const u16* __restrict__ Oldt,
                                                  u16* __restrict__ Outt,
                                                  const float* __restrict__ alphap,
                                                  int mode) {
    __shared__ u16 As[2][32 * 128];              // 2 x (32 rows x K=128), 16KB

    const int tid  = threadIdx.x;
    const int lane = tid & 31;
    const int wave = tid >> 5;                   // 0..7 -> col tile
    const int r0   = blockIdx.x * 32;            // output row base (32 rows)
    const int col0 = wave * 16;
    const int n16  = lane & 15;
    const int hiK  = lane >> 4;                  // 0: low-K half, 1: high-K half

    // staging map: each thread copies 32B (two us8) of the 32x128 A strip
    const int srow = tid >> 3;                   // 0..31
    const int scol = (tid & 7) * 16;             // 0,16,...,112
    const u16* ag  = adjB + (size_t)(r0 + srow) * NN + scol;
    const int sdst = srow * 128 + scol;

    const u16* brow = Bt + (size_t)(col0 + n16) * NN + hiK * 16;

    v8f acc0 = {0.f, 0.f, 0.f, 0.f, 0.f, 0.f, 0.f, 0.f};
    v8f acc1 = {0.f, 0.f, 0.f, 0.f, 0.f, 0.f, 0.f, 0.f};

    // prologue: stage 0 into buffer 0
    {
        us8 a0 = *(const us8*)(ag);
        us8 a1 = *(const us8*)(ag + 8);
        *(us8*)&As[0][sdst]     = a0;
        *(us8*)&As[0][sdst + 8] = a1;
    }
    __syncthreads();

    for (int k0 = 0; k0 < NN; k0 += 128) {
        const int buf   = (k0 >> 7) & 1;
        const int knext = (k0 + 128 < NN) ? (k0 + 128) : k0;  // last: dummy
        // issue next-stage A loads early; waited on only at the ds_store below
        us8 na0 = *(const us8*)(ag + knext);
        us8 na1 = *(const us8*)(ag + knext + 8);

#pragma unroll
        for (int kk = 0; kk < 4; ++kk) {
            // A fragment: lane<16 -> K 0..15; lane>=16 -> 16..31
            const int abase = kk * 32 + hiK * 16;
            BF16x16 a0, a1, b;
            // B fragment: lane<16 -> K 0..15 of its column; lane>=16 -> 16..31
            const u16* bp = brow + k0 + kk * 32;
            b.h.lo  = *(const us8*)bp;
            b.h.hi  = *(const us8*)(bp + 8);
            a0.h.lo = *(const us8*)&As[buf][n16 * 128 + abase];
            a0.h.hi = *(const us8*)&As[buf][n16 * 128 + abase + 8];
            a1.h.lo = *(const us8*)&As[buf][(16 + n16) * 128 + abase];
            a1.h.hi = *(const us8*)&As[buf][(16 + n16) * 128 + abase + 8];

            acc0 = __builtin_amdgcn_wmma_f32_16x16x32_bf16(
                       false, a0.v, false, b.v, (short)0, acc0, false, false);
            acc1 = __builtin_amdgcn_wmma_f32_16x16x32_bf16(
                       false, a1.v, false, b.v, (short)0, acc1, false, false);
        }

        // store next stage into the other buffer, then one barrier per stage
        *(us8*)&As[buf ^ 1][sdst]     = na0;
        *(us8*)&As[buf ^ 1][sdst + 8] = na1;
        __syncthreads();
    }

    // C/D layout: lane column = n16, VGPR r = row hiK*8 + r -> contiguous in
    // the transposed output buffer: one 16B store (and one 16B load for ODE).
    const float c1 = alphap[0] * 0.1f;           // dt*alpha, dt = 1/10
    const float c0 = 1.0f - c1;
    const size_t obase = (size_t)(col0 + n16) * NN + (size_t)r0 + hiK * 8;
    epi_store(acc0, Outt, Oldt, obase,      mode, c0, c1);   // rows r0..r0+15
    epi_store(acc1, Outt, Oldt, obase + 16, mode, c0, c1);   // rows r0+16..+31
}

// ---------------------------------------------------------------------------
// P1^T = bf16( x @ W1 + b1 )^T   (x fp32 row-major [NN,DIN])
// ---------------------------------------------------------------------------
__global__ __launch_bounds__(128) void k_proj1(const float* __restrict__ x,
                                               const float* __restrict__ W1,
                                               const float* __restrict__ b1,
                                               u16* __restrict__ Outt) {
    __shared__ float xr[DIN];
    const int row = blockIdx.x, t = threadIdx.x;
    xr[t]       = x[(size_t)row * DIN + t];
    xr[t + 128] = x[(size_t)row * DIN + 128 + t];
    __syncthreads();
    float s = b1[t];
#pragma unroll 4
    for (int k = 0; k < DIN; ++k) s = fmaf(xr[k], W1[k * HID + t], s);
    Outt[(size_t)t * NN + row] = f2bf(s);
}

// ---------------------------------------------------------------------------
// P2^T = bf16( h @ W2 + b2 )^T   (h given bf16-transposed [HID][NN])
// ---------------------------------------------------------------------------
__global__ __launch_bounds__(128) void k_proj2(const u16* __restrict__ Ht,
                                               const float* __restrict__ W2,
                                               const float* __restrict__ b2,
                                               u16* __restrict__ Outt) {
    __shared__ float hr[HID];
    const int row = blockIdx.x, t = threadIdx.x;
    hr[t] = bf2f(Ht[(size_t)t * NN + row]);
    __syncthreads();
    float s = b2[t];
#pragma unroll 4
    for (int k = 0; k < HID; ++k) s = fmaf(hr[k], W2[k * HID + t], s);
    Outt[(size_t)t * NN + row] = f2bf(s);
}

// ---------------------------------------------------------------------------
// Decoder: softmax( relu(h@W3+b3) @ W4 + b4 ), one block per row.
// ---------------------------------------------------------------------------
__global__ __launch_bounds__(128) void k_decoder(const u16* __restrict__ Ht,
                                                 const float* __restrict__ W3,
                                                 const float* __restrict__ b3,
                                                 const float* __restrict__ W4,
                                                 const float* __restrict__ b4,
                                                 float* __restrict__ out) {
    __shared__ float hr[HID];
    __shared__ float u[HID];
    __shared__ float o[DOUT];
    const int row = blockIdx.x, t = threadIdx.x;

    hr[t] = bf2f(Ht[(size_t)t * NN + row]);
    __syncthreads();

    float s = b3[t];
#pragma unroll 4
    for (int k = 0; k < HID; ++k) s = fmaf(hr[k], W3[k * HID + t], s);
    u[t] = fmaxf(s, 0.0f);
    __syncthreads();

    if (t < DOUT) {
        float s2 = b4[t];
#pragma unroll 4
        for (int j = 0; j < HID; ++j) s2 = fmaf(u[j], W4[j * DOUT + t], s2);
        o[t] = s2;
    }
    __syncthreads();

    if (t < DOUT) {
        float m = -3.4e38f;
        for (int j = 0; j < DOUT; ++j) m = fmaxf(m, o[j]);
        float sum = 0.0f;
        for (int j = 0; j < DOUT; ++j) sum += __expf(o[j] - m);
        out[(size_t)row * DOUT + t] = __expf(o[t] - m) / sum;
    }
}

// ---------------------------------------------------------------------------
extern "C" void kernel_launch(void* const* d_in, const int* in_sizes, int n_in,
                              void* d_out, int out_size, void* d_ws, size_t ws_size,
                              hipStream_t stream) {
    const float* x     = (const float*)d_in[0];
    // d_in[1] = edge_index (unused by the reference computation)
    const float* adj   = (const float*)d_in[2];
    const float* W1    = (const float*)d_in[3];
    const float* b1    = (const float*)d_in[4];
    const float* W2    = (const float*)d_in[5];
    const float* b2    = (const float*)d_in[6];
    const float* alpha = (const float*)d_in[7];
    const float* W3    = (const float*)d_in[8];
    const float* b3    = (const float*)d_in[9];
    const float* W4    = (const float*)d_in[10];
    const float* b4    = (const float*)d_in[11];
    float* out = (float*)d_out;

    u16* adjB = (u16*)d_ws;                                  // 128 MB
    u16* T0   = adjB + (size_t)NN * NN;                      // 2 MB
    u16* T1   = T0 + (size_t)HID * NN;                       // 2 MB

    // 1) adj -> bf16 (L2-resident working set for all 12 GEMMs)
    k_conv_adj<<<(NN * (size_t)NN) / (256 * 4), 256, 0, stream>>>(adj, adjB);

    // 2) P1^T = (x@W1+b1)^T
    k_proj1<<<NN, 128, 0, stream>>>(x, W1, b1, T0);

    // 3) h = relu(adj @ P1)          -> T1
    k_gemm_adj<<<NN / 32, 256, 0, stream>>>(adjB, T0, T0, T1, alpha, 1);

    // 4) P2^T = (h@W2+b2)^T          -> T0
    k_proj2<<<NN, 128, 0, stream>>>(T1, W2, b2, T0);

    // 5) h0 = adj @ P2               -> T1
    k_gemm_adj<<<NN / 32, 256, 0, stream>>>(adjB, T0, T0, T1, alpha, 0);

    // 6) 10 Euler steps: h' = (1-dt*a)h + dt*a*(A h), ping-pong T1<->T0
    u16* cur = T1; u16* nxt = T0;
    for (int s = 0; s < 10; ++s) {
        k_gemm_adj<<<NN / 32, 256, 0, stream>>>(adjB, cur, cur, nxt, alpha, 2);
        u16* tmp = cur; cur = nxt; nxt = tmp;
    }
    // 10 swaps -> final h back in T1 (cur == T1)

    // 7) decoder + softmax
    k_decoder<<<NN, 128, 0, stream>>>(cur, W3, b3, W4, b4, out);
}
